// NodeNetwork_14233521619351
// MI455X (gfx1250) — matrix operand order repaired and checked by reading, not compiled, not run on hardware
//
#include <hip/hip_runtime.h>

typedef _Float16 v16h __attribute__((ext_vector_type(16)));
typedef _Float16 v8h  __attribute__((ext_vector_type(8)));
typedef float    v8f  __attribute__((ext_vector_type(8)));

#define DHID 128

__device__ __forceinline__ float fast_tanh(float xx) {
    // tanh(x) = 1 - 2/(exp(2x)+1); v_exp_f32 + v_rcp_f32, ~1 ulp each.
    float e = __builtin_amdgcn_exp2f(xx * 2.885390081777927f); // 2*log2(e)
    return 1.0f - 2.0f * __builtin_amdgcn_rcpf(e + 1.0f);
}

// ---------------------------------------------------------------------------
// Weight pre-conversion: f32 -> f16 (done every call; deterministic & cheap)
// ---------------------------------------------------------------------------
__global__ void cvt_weights(const float* __restrict__ W0, const float* __restrict__ W,
                            _Float16* __restrict__ W0h, _Float16* __restrict__ Wh,
                            int n0, int n1) {
    int i = blockIdx.x * blockDim.x + threadIdx.x;
    if (i < n0)            W0h[i]      = (_Float16)W0[i];
    else if (i < n0 + n1)  Wh[i - n0]  = (_Float16)W[i - n0];
}

// ---------------------------------------------------------------------------
// Edge aggregation: one wave (32 lanes) per edge, 4 columns per lane.
//   mi[end]   += e * x[start]
//   mo[start] += e * x[end]
// Rows are 512B contiguous -> fully coalesced; atomics land in L2 (51MB
// accumulators << 192MB L2).
// ---------------------------------------------------------------------------
__global__ void edge_scatter(const float* __restrict__ x, const float* __restrict__ e,
                             const int* __restrict__ ei,
                             float* __restrict__ mi, float* __restrict__ mo,
                             int nEdges) {
    int gid  = blockIdx.x * blockDim.x + threadIdx.x;
    int edge = gid >> 5;
    if (edge >= nEdges) return;
    int lane = gid & 31;
    int s = ei[edge];
    int t = ei[nEdges + edge];
    float w = e[edge];
    int c0 = lane * 4;

    float4 xs = *reinterpret_cast<const float4*>(x + (size_t)s * DHID + c0);
    float4 xt = *reinterpret_cast<const float4*>(x + (size_t)t * DHID + c0);

    float* mid = mi + (size_t)t * DHID + c0;
    float* mod = mo + (size_t)s * DHID + c0;
    unsafeAtomicAdd(mid + 0, w * xs.x);
    unsafeAtomicAdd(mid + 1, w * xs.y);
    unsafeAtomicAdd(mid + 2, w * xs.z);
    unsafeAtomicAdd(mid + 3, w * xs.w);
    unsafeAtomicAdd(mod + 0, w * xt.x);
    unsafeAtomicAdd(mod + 1, w * xt.y);
    unsafeAtomicAdd(mod + 2, w * xt.z);
    unsafeAtomicAdd(mod + 3, w * xt.w);
}

// ---------------------------------------------------------------------------
// LayerNorm + tanh directly on the WMMA C-fragment layout.
// C layout (16x16 f32 D, 8 regs): element (M = r + 8*(lane>>4), N = lane&15),
// tile column = 16*nt + N. Row reduction: 8 per-lane partials (over nt) then
// shfl_xor over masks 1,2,4,8 (stays inside each 16-lane half-group).
// ---------------------------------------------------------------------------
__device__ __forceinline__ void ln_tanh_store(
    v8f C[8], const float* __restrict__ bias, const float* __restrict__ gamma,
    const float* __restrict__ beta, int laneM, int laneH,
    _Float16* ldsDst,            // non-null: write f16 activations for next layer
    float* gOut, int row0, int nNodes)  // ldsDst==nullptr: final f32 global store
{
    float bi[8], ga[8], bb[8];
#pragma unroll
    for (int nt = 0; nt < 8; ++nt) {
        int col = nt * 16 + laneM;
        bi[nt] = bias[col]; ga[nt] = gamma[col]; bb[nt] = beta[col];
    }
#pragma unroll
    for (int r = 0; r < 8; ++r) {
        float v[8];
        float s = 0.f, q = 0.f;
#pragma unroll
        for (int nt = 0; nt < 8; ++nt) {
            v[nt] = C[nt][r] + bi[nt];
            s += v[nt];
            q += v[nt] * v[nt];
        }
#pragma unroll
        for (int m = 1; m < 16; m <<= 1) {
            s += __shfl_xor(s, m, 32);
            q += __shfl_xor(q, m, 32);
        }
        float mean = s * (1.0f / DHID);
        float var  = q * (1.0f / DHID) - mean * mean;
        float rs   = __builtin_amdgcn_rsqf(var + 1e-5f);
        int row = r + 8 * laneH;
#pragma unroll
        for (int nt = 0; nt < 8; ++nt) {
            float t = fast_tanh((v[nt] - mean) * rs * ga[nt] + bb[nt]);
            int col = nt * 16 + laneM;
            if (ldsDst) {
                ldsDst[row * DHID + col] = (_Float16)t;
            } else {
                int node = row0 + row;
                if (node < nNodes) gOut[(size_t)node * DHID + col] = t;
            }
        }
    }
}

// ---------------------------------------------------------------------------
// Fused 4-layer MLP. One wave per 16-node tile; 8 waves (8 tiles) per block.
// Layer 0: K=384 (concat [mi,mo,x]) in 12 K-chunks of 32.
// Layers 1..3: K=128 from per-wave LDS f16 tile, 4 K-chunks.
// A-frag (16x32 f16): lane holds row M=lane&15; halfs j=0..7 -> K=8*(lane>>4)+j,
//                     j=8..15 -> K=16+8*(lane>>4)+(j-8).
// B-frag (32x16 f16): lane holds W column N=lane&15 (within tile); halfs j -> same K as A-frag.
// ---------------------------------------------------------------------------
__global__ void __launch_bounds__(256) mlp_fused(
    const float* __restrict__ mi, const float* __restrict__ mo,
    const float* __restrict__ x,
    const _Float16* __restrict__ W0h, const _Float16* __restrict__ Wh,
    const float* __restrict__ b0, const float* __restrict__ g0,
    const float* __restrict__ be0,
    const float* __restrict__ bL3, const float* __restrict__ gL3,
    const float* __restrict__ beL3,
    float* __restrict__ out, int nNodes)
{
    __shared__ __align__(16) _Float16 ldsH[8][16 * DHID];  // 32KB: per-wave f16 tiles

    const int tid   = threadIdx.x;
    const int wave  = tid >> 5;
    const int lane  = tid & 31;
    const int laneM = lane & 15;
    const int laneH = lane >> 4;
    const int tile  = blockIdx.x * 8 + wave;
    const int row0  = tile * 16;

    _Float16* myLds = ldsH[wave];

    // Clamped node for this lane's A-matrix row (out-of-range waves compute
    // on node 0 and simply skip the final store -> uniform barriers, EXEC=all-1s).
    int ndA = row0 + laneM;
    if (ndA >= nNodes) ndA = 0;

    v8f C[8];

    // ---------------- Layer 0: [mi|mo|x](16x384) @ W0(384x128) ----------------
#pragma unroll
    for (int nt = 0; nt < 8; ++nt) C[nt] = {};

    for (int c = 0; c < 12; ++c) {
        int off = c * 32;
        const float* src;
        if (off < 128)      src = mi + (size_t)ndA * DHID + off;
        else if (off < 256) src = mo + (size_t)ndA * DHID + (off - 128);
        else                src = x  + (size_t)ndA * DHID + (off - 256);

        const int k0 = laneH * 8;
        v16h A;
#pragma unroll
        for (int j = 0; j < 8; ++j) A[j]     = (_Float16)src[k0 + j];
#pragma unroll
        for (int j = 0; j < 8; ++j) A[8 + j] = (_Float16)src[16 + k0 + j];

        const _Float16* wcol = W0h + (size_t)(c * 32) * DHID + laneM;
#pragma unroll
        for (int nt = 0; nt < 8; ++nt) {
            v16h B;
#pragma unroll
            for (int j = 0; j < 8; ++j) { B[j] = wcol[(size_t)(k0 + j) * DHID + nt * 16]; B[8 + j] = wcol[(size_t)(16 + k0 + j) * DHID + nt * 16]; }
            C[nt] = __builtin_amdgcn_wmma_f32_16x16x32_f16(
                false, A, false, B, (short)0, C[nt], false, false);
        }
    }

    ln_tanh_store(C, b0, g0, be0, laneM, laneH, myLds, nullptr, row0, nNodes);
    __syncthreads();

    // ---------------- Layers 1..3: h(16x128) @ W[L](128x128) ----------------
    for (int L = 0; L < 3; ++L) {
#pragma unroll
        for (int nt = 0; nt < 8; ++nt) C[nt] = {};

        const _Float16* Wl = Wh + (size_t)L * DHID * DHID;
#pragma unroll
        for (int c = 0; c < 4; ++c) {
            const int k0 = c * 32 + laneH * 8;
            v8h lo = *reinterpret_cast<const v8h*>(myLds + laneM * DHID + k0);
            v8h hi = *reinterpret_cast<const v8h*>(myLds + laneM * DHID + k0 + 16);
            v16h A;
#pragma unroll
            for (int j = 0; j < 8; ++j) { A[j] = lo[j]; A[8 + j] = hi[j]; }

            const _Float16* wcol = Wl + laneM;
#pragma unroll
            for (int nt = 0; nt < 8; ++nt) {
                v16h B;
#pragma unroll
                for (int j = 0; j < 8; ++j) { B[j] = wcol[(size_t)(k0 + j) * DHID + nt * 16]; B[8 + j] = wcol[(size_t)(16 + k0 + j) * DHID + nt * 16]; }
                C[nt] = __builtin_amdgcn_wmma_f32_16x16x32_f16(
                    false, A, false, B, (short)0, C[nt], false, false);
            }
        }

        __syncthreads();  // all A-reads of myLds done before LN overwrites it

        const float* bL  = bL3  + L * DHID;
        const float* gL  = gL3  + L * DHID;
        const float* beL = beL3 + L * DHID;
        if (L < 2)
            ln_tanh_store(C, bL, gL, beL, laneM, laneH, myLds, nullptr, row0, nNodes);
        else
            ln_tanh_store(C, bL, gL, beL, laneM, laneH, nullptr, out, row0, nNodes);
        __syncthreads();
    }
}

// ---------------------------------------------------------------------------
// Host launch
// ---------------------------------------------------------------------------
extern "C" void kernel_launch(void* const* d_in, const int* in_sizes, int n_in,
                              void* d_out, int out_size, void* d_ws, size_t ws_size,
                              hipStream_t stream) {
    (void)n_in; (void)out_size; (void)ws_size;
    const float* x   = (const float*)d_in[0];
    const float* e   = (const float*)d_in[1];
    const int*   ei  = (const int*)d_in[2];
    const float* W0  = (const float*)d_in[3];
    const float* b0  = (const float*)d_in[4];
    const float* g0  = (const float*)d_in[5];
    const float* be0 = (const float*)d_in[6];
    const float* W   = (const float*)d_in[7];
    const float* b   = (const float*)d_in[8];
    const float* g   = (const float*)d_in[9];
    const float* be  = (const float*)d_in[10];
    float* out = (float*)d_out;

    const int nNodes = in_sizes[0] / DHID;   // 100000
    const int nEdges = in_sizes[1];          // 1.6M
    const int nW0    = in_sizes[3];          // 384*128
    const int nW     = in_sizes[7];          // 3*128*128

    char* ws = (char*)d_ws;
    size_t szMi = (size_t)nNodes * DHID * sizeof(float);   // 51.2 MB
    float*    mi  = (float*)ws;
    float*    mo  = (float*)(ws + szMi);
    _Float16* W0h = (_Float16*)(ws + 2 * szMi);
    _Float16* Wh  = W0h + nW0;

    // Zero the segment-sum accumulators (graph-capture-safe async memset).
    hipMemsetAsync(mi, 0, 2 * szMi, stream);

    // f32 -> f16 weight conversion.
    int nCvt = nW0 + nW;
    cvt_weights<<<(nCvt + 255) / 256, 256, 0, stream>>>(W0, W, W0h, Wh, nW0, nW);

    // Edge aggregation: one wave per edge -> 8 edges per 256-thread block.
    edge_scatter<<<(nEdges + 7) / 8, 256, 0, stream>>>(x, e, ei, mi, mo, nEdges);

    // Fused MLP: 16-node tile per wave, 8 tiles per block.
    int tiles  = (nNodes + 15) / 16;
    int blocks = (tiles + 7) / 8;
    mlp_fused<<<blocks, 256, 0, stream>>>(mi, mo, x, W0h, Wh,
                                          b0, g0, be0, b, g, be, out, nNodes);
}
